// Pool_SAGEConv_79920751444223
// MI455X (gfx1250) — matrix operand order repaired and checked by reading, not compiled, not run on hardware
//
#include <hip/hip_runtime.h>
#include <hip/hip_bf16.h>

typedef __attribute__((ext_vector_type(16))) _Float16 v16h;
typedef __attribute__((ext_vector_type(8)))  float    v8f;

#define CH 64
#define EDGES_PER_BLOCK 128
#define NODES_PER_BLOCK 128

// ---- packing helper: two f32 -> packed f16 pair (v_cvt_pk_f16_f32) -------
__device__ __forceinline__ unsigned pack2(float a, float b) {
    union { _Float16 h[2]; unsigned u; } u;
    u.h[0] = (_Float16)a;
    u.h[1] = (_Float16)b;
    return u.u;
}

// ---- WMMA fragment helpers (ISA 7.12.2, wave32, 16x16x32 f16) ------------

// A-matrix 16x32 (16-bit): lane l -> M = l%16, hi = l/16.
// VGPR v<4 : K = 2v+j + hi*8 ; v>=4 : K = 16 + 2(v-4)+j + hi*8
__device__ __forceinline__ int kmapA(int v, int j, int hi) {
    int base = (v < 4) ? (2 * v) : (16 + 2 * (v - 4));
    return base + j + hi * 8;
}
// B-matrix 32x16 (16-bit): lane l -> N = l%16, hi = l/16, K same as kmapA
__device__ __forceinline__ int kmapB(int v, int j, int hi) {
    return ((v < 4) ? (2 * v) : (16 + 2 * (v - 4))) + j + hi * 8;
}

__device__ __forceinline__ v16h load_A_frag(const _Float16* lds, int ld, int row0, int k0) {
    int lane = threadIdx.x & 31;
    int m    = lane & 15;
    int hi   = lane >> 4;
    v16h a;
#pragma unroll
    for (int v = 0; v < 8; ++v)
#pragma unroll
        for (int j = 0; j < 2; ++j)
            a[2 * v + j] = lds[(row0 + m) * ld + k0 + kmapA(v, j, hi)];
    return a;
}

// B fragment from weight matrix stored row-major as W[n][k] (ld = K):
// B[k][n] = W[n][k]  (implements y = x @ W^T).
__device__ __forceinline__ v16h load_B_frag(const _Float16* w, int ld, int n0, int k0) {
    int lane = threadIdx.x & 31;
    int n    = lane & 15;
    int hi   = lane >> 4;
    v16h b;
#pragma unroll
    for (int v = 0; v < 8; ++v)
#pragma unroll
        for (int j = 0; j < 2; ++j)
            b[2 * v + j] = w[(n0 + n) * ld + k0 + kmapB(v, j, hi)];
    return b;
}

__device__ __forceinline__ v8f wmma_f16(v16h a, v16h b, v8f c) {
    return __builtin_amdgcn_wmma_f32_16x16x32_f16(
        false, a, false, b, (short)0, c, false, false);
}

// ---- Kernel 0: zero the segment-max accumulator (float4) -----------------
__global__ __launch_bounds__(256) void zero_agg(float4* __restrict__ p, long long n4) {
    long long i = (long long)blockIdx.x * blockDim.x + threadIdx.x;
    if (i < n4) p[i] = make_float4(0.f, 0.f, 0.f, 0.f);
}

// ---- Kernel 1: per-edge pool linear + scatter-max ------------------------
// One block = 128 messages; wave w handles messages [16w, 16w+16).
// No explicit ReLU: agg starts at 0 and atomicMax on IEEE bits (signed int
// compare) never takes a negative float, so result == segment_max(relu(v))∪{0}.
__global__ __launch_bounds__(256) void edge_pool(
    const float* __restrict__ x,            // [N][64]
    const long long* __restrict__ ei,       // [2][E]
    const float* __restrict__ ew,           // [E]
    const float* __restrict__ pool_w,       // [64][64]  (out, in)
    const float* __restrict__ pool_b,       // [64]
    const float* __restrict__ coefp,        // [1]
    int* __restrict__ agg_i,                // [N][64] fp32 bits, pre-zeroed
    long long E, long long numMsgs, long long dummyBase)
{
    __shared__ _Float16 sEF[EDGES_PER_BLOCK * CH];   // 16 KB edge features
    __shared__ _Float16 sW[CH * CH];                 //  8 KB pool_w as f16
    __shared__ float    sB[CH];
    __shared__ int      sDst[EDGES_PER_BLOCK];       // dest node (-1 => skip)

    const int t = threadIdx.x;
    const float coef = coefp[0];

    // stage pool_w as packed f16 (uint4 = 8 halves per store)
    for (int i = t; i < (CH * CH) / 8; i += 256) {
        float4 lo = *(const float4*)(pool_w + i * 8);
        float4 hi = *(const float4*)(pool_w + i * 8 + 4);
        uint4 q;
        q.x = pack2(lo.x, lo.y);  q.y = pack2(lo.z, lo.w);
        q.z = pack2(hi.x, hi.y);  q.w = pack2(hi.z, hi.w);
        ((uint4*)sW)[i] = q;
    }
    if (t < CH) sB[t] = pool_b[t];

    // stage edge features: 2 threads per message, 32 channels each
    const long long base = (long long)blockIdx.x * EDGES_PER_BLOCK;
    const int eL   = t >> 1;
    const int half = t & 1;
    const long long m = base + eL;
    long long s = 0, d = 0;
    float w = 0.0f;
    const int valid = (m < numMsgs);
    if (valid) {
        if (m < E) { s = ei[m];            d = ei[E + m];   w = ew[m]; }
        else       { long long mm = m - E; s = ei[E + mm];  d = ei[mm]; w = ew[mm]; }
    }
    const float scale = 1.0f + coef * w;
    const int c0 = half * 32;
    uint4* efDst = (uint4*)(sEF + eL * CH + c0);
    if (valid) {
        const float* xs = x + s * CH + c0;
#pragma unroll
        for (int i = 0; i < 4; ++i) {
            float4 lo = *(const float4*)(xs + i * 8);
            float4 hi = *(const float4*)(xs + i * 8 + 4);
            uint4 q;
            q.x = pack2(lo.x * scale, lo.y * scale);
            q.y = pack2(lo.z * scale, lo.w * scale);
            q.z = pack2(hi.x * scale, hi.y * scale);
            q.w = pack2(hi.z * scale, hi.w * scale);
            efDst[i] = q;
        }
    } else {
        uint4 z = make_uint4(0u, 0u, 0u, 0u);
#pragma unroll
        for (int i = 0; i < 4; ++i) efDst[i] = z;
    }
    if (half == 0) {
        sDst[eL] = ((!valid) || (s == d)) ? -1 : (int)d;
    }
    __syncthreads();

    // compute: 16 edges x 64 outputs per wave (EXEC full for every WMMA)
    const int wave = t >> 5;
    const int lane = t & 31;
    const int er0  = wave * 16;
    const int nsub = lane & 15;
    const int hi   = lane >> 4;

    v16h a0 = load_A_frag(sEF, CH, er0, 0);
    v16h a1 = load_A_frag(sEF, CH, er0, 32);

    // Branchless commit targets: this lane owns rows er0+hi*8 .. +7.
    long long rowIdx[8];
    const int rbase = er0 + hi * 8;
#pragma unroll
    for (int r = 0; r < 8; ++r) {
        int dn = sDst[rbase + r];
        rowIdx[r] = (dn < 0) ? (dummyBase + nsub) : ((long long)dn * CH + nsub);
    }

    // All 4 output tiles first (independent WMMA chains -> no D->commit stall
    // between tiles), then one big atomic commit.
    v8f acc[4];
#pragma unroll
    for (int nt = 0; nt < 4; ++nt) {
        const int n0 = nt * 16;
        v16h b0 = load_B_frag(sW, CH, n0, 0);
        v16h b1 = load_B_frag(sW, CH, n0, 32);
        v8f c;
        const float bias = sB[n0 + nsub];
#pragma unroll
        for (int r = 0; r < 8; ++r) c[r] = bias;
        c = wmma_f16(a0, b0, c);
        acc[nt] = wmma_f16(a1, b1, c);
    }

#pragma unroll
    for (int nt = 0; nt < 4; ++nt)
#pragma unroll
        for (int r = 0; r < 8; ++r)
            atomicMax(agg_i + (rowIdx[r] + nt * 16), __float_as_int(acc[nt][r]));
}

// ---- Kernel 2: out = relu([x | agg] @ final_w^T + final_b) ---------------
__global__ __launch_bounds__(256) void final_linear(
    const float* __restrict__ x,        // [N][64]
    const float* __restrict__ agg,      // [N][64]
    const float* __restrict__ fw,       // [64][128]
    const float* __restrict__ fb,       // [64]
    float* __restrict__ out,            // [N][64]
    int nNodes)
{
    __shared__ _Float16 sH[NODES_PER_BLOCK * 2 * CH]; // 32 KB [node][128]
    __shared__ _Float16 sW[CH * 2 * CH];              // 16 KB
    __shared__ float    sB[CH];

    const int t = threadIdx.x;
    for (int i = t; i < (CH * 2 * CH) / 8; i += 256) {
        float4 lo = *(const float4*)(fw + i * 8);
        float4 hi = *(const float4*)(fw + i * 8 + 4);
        uint4 q;
        q.x = pack2(lo.x, lo.y);  q.y = pack2(lo.z, lo.w);
        q.z = pack2(hi.x, hi.y);  q.w = pack2(hi.z, hi.w);
        ((uint4*)sW)[i] = q;
    }
    if (t < CH) sB[t] = fb[t];

    const int base = blockIdx.x * NODES_PER_BLOCK;
    const bool fullBlock = (base + NODES_PER_BLOCK) <= nNodes;
    const int row  = t >> 1;
    const int half = t & 1;                // 0 -> x, 1 -> agg
    const int node = base + row;
    const int c0   = half * CH;
    uint4* hDst = (uint4*)(sH + row * 2 * CH + c0);
    if (fullBlock || node < nNodes) {
        const float* src = (half == 0) ? (x + (long long)node * CH)
                                       : (agg + (long long)node * CH);
#pragma unroll
        for (int i = 0; i < 8; ++i) {
            float4 lo = *(const float4*)(src + i * 8);
            float4 hi = *(const float4*)(src + i * 8 + 4);
            uint4 q;
            q.x = pack2(lo.x, lo.y);  q.y = pack2(lo.z, lo.w);
            q.z = pack2(hi.x, hi.y);  q.w = pack2(hi.z, hi.w);
            hDst[i] = q;
        }
    } else {
        uint4 z = make_uint4(0u, 0u, 0u, 0u);
#pragma unroll
        for (int i = 0; i < 8; ++i) hDst[i] = z;
    }
    __syncthreads();

    const int wave = t >> 5;
    const int lane = t & 31;
    const int r0   = wave * 16;
    const int nsub = lane & 15;
    const int hi   = lane >> 4;

    v16h a[4];
#pragma unroll
    for (int kc = 0; kc < 4; ++kc) a[kc] = load_A_frag(sH, 2 * CH, r0, kc * 32);

    // compute all 4 output tiles, then store
    v8f acc[4];
#pragma unroll
    for (int nt = 0; nt < 4; ++nt) {
        v8f c;
        const float bias = sB[nt * 16 + nsub];
#pragma unroll
        for (int r = 0; r < 8; ++r) c[r] = bias;
#pragma unroll
        for (int kc = 0; kc < 4; ++kc) {
            v16h b = load_B_frag(sW, 2 * CH, nt * 16, kc * 32);
            c = wmma_f16(a[kc], b, c);
        }
        acc[nt] = c;
    }

    const int myRow0 = base + r0 + hi * 8;   // first of this lane's 8 rows
    float* o0 = out + (long long)myRow0 * CH + nsub;
    if (fullBlock) {
#pragma unroll
        for (int nt = 0; nt < 4; ++nt)
#pragma unroll
            for (int r = 0; r < 8; ++r) {
                float v = acc[nt][r];
                o0[(long long)r * CH + nt * 16] = v > 0.0f ? v : 0.0f;
            }
    } else {
#pragma unroll
        for (int nt = 0; nt < 4; ++nt)
#pragma unroll
            for (int r = 0; r < 8; ++r)
                if (myRow0 + r < nNodes) {
                    float v = acc[nt][r];
                    o0[(long long)r * CH + nt * 16] = v > 0.0f ? v : 0.0f;
                }
    }
}

extern "C" void kernel_launch(void* const* d_in, const int* in_sizes, int n_in,
                              void* d_out, int out_size, void* d_ws, size_t ws_size,
                              hipStream_t stream) {
    const float*     x      = (const float*)d_in[0];
    const long long* ei     = (const long long*)d_in[1];   // int64 [2][E]
    const float*     ew     = (const float*)d_in[2];
    const float*     pool_w = (const float*)d_in[3];
    const float*     pool_b = (const float*)d_in[4];
    const float*     fw     = (const float*)d_in[5];
    const float*     fb     = (const float*)d_in[6];
    const float*     coefp  = (const float*)d_in[7];

    const int       nNodes  = in_sizes[0] / CH;
    const long long E       = (long long)in_sizes[2];
    const long long numMsgs = 2 * E;

    float* agg = (float*)d_ws;                       // [N][64] fp32 + 64 dummy
    float* out = (float*)d_out;
    const long long dummyBase = (long long)nNodes * CH;

    const long long aggN = dummyBase + CH;           // include dummy bucket
    const long long n4   = (aggN + 3) / 4;
    int zb = (int)((n4 + 255) / 256);
    zero_agg<<<zb, 256, 0, stream>>>((float4*)agg, n4);

    int eb = (int)((numMsgs + EDGES_PER_BLOCK - 1) / EDGES_PER_BLOCK);
    edge_pool<<<eb, 256, 0, stream>>>(x, ei, ew, pool_w, pool_b, coefp,
                                      (int*)agg, E, numMsgs, dummyBase);

    int nb = (nNodes + NODES_PER_BLOCK - 1) / NODES_PER_BLOCK;
    final_linear<<<nb, 256, 0, stream>>>(x, agg, fw, fb, out, nNodes);
}
